// SparseAutoencoder_31911607009826
// MI455X (gfx1250) — matrix operand order repaired and checked by reading, not compiled, not run on hardware
//
#include <hip/hip_runtime.h>

#define D_MODEL   2048
#define D_HIDDEN  8192
#define BATCH     8192
#define TOPK      64

typedef __attribute__((ext_vector_type(16))) __bf16         v16bf;
typedef __attribute__((ext_vector_type(2)))  __bf16         v2bf;
typedef __attribute__((ext_vector_type(8)))  float          v8f;
typedef __attribute__((ext_vector_type(8)))  unsigned short v8us;

union FragU { v16bf v; v8us h[2]; };

__device__ __forceinline__ unsigned short f32_to_bf16(float f) {
    unsigned u = __float_as_uint(f);
    unsigned r = u + 0x7FFFu + ((u >> 16) & 1u);   // round to nearest even
    return (unsigned short)(r >> 16);
}

__device__ __forceinline__ unsigned pack2_bf16(float a, float b) {
#if __has_builtin(__builtin_amdgcn_cvt_pk_bf16_f32)
    v2bf p = __builtin_amdgcn_cvt_pk_bf16_f32(a, b);
    return __builtin_bit_cast(unsigned, p);
#else
    return (unsigned)f32_to_bf16(a) | ((unsigned)f32_to_bf16(b) << 16);
#endif
}

// ---------------------------------------------------------------------------
// Encode: pre[B,H] = (x - b_pre) @ W_enc^T + b_enc   via bf16 WMMA, f32 accum
// Block tile 128x128, 8 waves of wave32, each wave 64x32 (4x2 accumulators),
// K-step 32, double-buffered LDS staging with packed fp32->bf16 convert.
// ---------------------------------------------------------------------------
#define BK 32
#define LDS_STRIDE 40   // 32 bf16 + 8 pad; row pitch 80B (16B aligned)

__global__ __launch_bounds__(256)
void encode_kernel(const float* __restrict__ x,
                   const float* __restrict__ W_enc,
                   const float* __restrict__ b_enc,
                   const float* __restrict__ b_pre,
                   float* __restrict__ pre_out) {
    __shared__ __align__(16) unsigned short As[2][128 * LDS_STRIDE];
    __shared__ __align__(16) unsigned short Bs[2][128 * LDS_STRIDE];

    const int tid   = threadIdx.x;
    const int lane  = tid & 31;
    const int wave  = tid >> 5;
    const int waveM = wave >> 2;           // 0..1  -> 64-row strip
    const int waveN = wave & 3;            // 0..3  -> 32-col strip
    const int blockM = blockIdx.y * 128;
    const int blockN = blockIdx.x * 128;
    const int lrow   = lane & 15;
    const int lhalf  = (lane < 16) ? 0 : 1;

    v8f acc[4][2] = {};

    float4 aReg[4], bReg[4];

    auto stageLoad = [&](int kbase) {
        #pragma unroll
        for (int q = 0; q < 4; ++q) {
            int l = tid + q * 256;          // 1024 float4 slots per matrix
            int r = l >> 3;                 // row 0..127
            int c = (l & 7) << 2;           // col 0,4,...,28
            float4 av = *(const float4*)(x + (size_t)(blockM + r) * D_MODEL + kbase + c);
            float4 pv = *(const float4*)(b_pre + kbase + c);
            av.x -= pv.x; av.y -= pv.y; av.z -= pv.z; av.w -= pv.w;
            aReg[q] = av;
            bReg[q] = *(const float4*)(W_enc + (size_t)(blockN + r) * D_MODEL + kbase + c);
        }
    };
    auto stageStore = [&](int buf) {
        #pragma unroll
        for (int q = 0; q < 4; ++q) {
            int l = tid + q * 256;
            int r = l >> 3;
            int c = (l & 7) << 2;
            uint2 pa;
            pa.x = pack2_bf16(aReg[q].x, aReg[q].y);
            pa.y = pack2_bf16(aReg[q].z, aReg[q].w);
            *(uint2*)(&As[buf][r * LDS_STRIDE + c]) = pa;   // one 8B ds_store
            uint2 pb;
            pb.x = pack2_bf16(bReg[q].x, bReg[q].y);
            pb.y = pack2_bf16(bReg[q].z, bReg[q].w);
            *(uint2*)(&Bs[buf][r * LDS_STRIDE + c]) = pb;
        }
    };

    stageLoad(0);
    stageStore(0);
    __syncthreads();

    const int NK = D_MODEL / BK;            // 64
    for (int kk = 0; kk < NK; ++kk) {
        const int cur = kk & 1;
        if (kk + 1 < NK) stageLoad((kk + 1) * BK);

        // A fragment (16-bit 16x32): lane<16 holds K 0..7 & 16..23, lane>=16 K 8..15 & 24..31
        FragU afrag[4], bfrag[2];
        #pragma unroll
        for (int i = 0; i < 4; ++i) {
            int row = waveM * 64 + i * 16 + lrow;
            const unsigned short* p = &As[cur][row * LDS_STRIDE + lhalf * 8];
            afrag[i].h[0] = *(const v8us*)p;
            afrag[i].h[1] = *(const v8us*)(p + 16);
        }
        // B fragment (32x16): lane<16 -> N=lane, K 0..7 & 16..23; lane>=16 -> N=lane-16, K 8..15 & 24..31
        #pragma unroll
        for (int j = 0; j < 2; ++j) {
            int col = waveN * 32 + j * 16 + lrow;
            const unsigned short* p = &Bs[cur][col * LDS_STRIDE + lhalf * 8];
            bfrag[j].h[0] = *(const v8us*)p;
            bfrag[j].h[1] = *(const v8us*)(p + 16);
        }
        #pragma unroll
        for (int i = 0; i < 4; ++i)
            #pragma unroll
            for (int j = 0; j < 2; ++j)
                acc[i][j] = __builtin_amdgcn_wmma_f32_16x16x32_bf16(
                    false, afrag[i].v, false, bfrag[j].v,
                    (short)0, acc[i][j], false, false);

        __syncthreads();
        if (kk + 1 < NK) { stageStore(1 - cur); __syncthreads(); }
    }

    // C/D layout: VGPR r -> M=r (lanes 0..15) / M=r+8 (lanes 16..31), N=lane%16
    #pragma unroll
    for (int i = 0; i < 4; ++i) {
        #pragma unroll
        for (int j = 0; j < 2; ++j) {
            int colg = blockN + waveN * 32 + j * 16 + lrow;
            float be = b_enc[colg];
            #pragma unroll
            for (int r = 0; r < 8; ++r) {
                int rowg = blockM + waveM * 64 + i * 16 + r + (lhalf ? 8 : 0);
                pre_out[(size_t)rowg * D_HIDDEN + colg] = acc[i][j][r] + be;
            }
        }
    }
}

// ---------------------------------------------------------------------------
// Per-row top-64 radix-select (order-preserving uint keys), deterministic
// tie-break by index; rewrites pre -> hidden in place (zeros + relu(vals)).
// ---------------------------------------------------------------------------
__global__ __launch_bounds__(256)
void topk_kernel(float* __restrict__ hidden) {
    const int row = blockIdx.x;
    float* p = hidden + (size_t)row * D_HIDDEN;

    __shared__ unsigned keys[D_HIDDEN];     // 32 KB
    __shared__ unsigned hist[256];
    __shared__ unsigned scn[257];
    __shared__ unsigned s_prefix, s_kwant;

    const int tid  = threadIdx.x;
    const int base = tid * 32;

    for (int i = 0; i < 32; ++i) {
        unsigned u = __float_as_uint(p[base + i]);
        keys[base + i] = (u & 0x80000000u) ? ~u : (u ^ 0x80000000u);
    }
    if (tid == 0) { s_prefix = 0u; s_kwant = TOPK; }
    __syncthreads();

    #pragma unroll
    for (int pass = 0; pass < 4; ++pass) {
        const int shift = 24 - pass * 8;
        const unsigned maskhi = (pass == 0) ? 0u : (0xFFFFFFFFu << (shift + 8));
        hist[tid] = 0u;
        __syncthreads();
        const unsigned prefix = s_prefix;
        for (int i = 0; i < 32; ++i) {
            unsigned key = keys[base + i];
            if ((key & maskhi) == prefix)
                atomicAdd(&hist[(key >> shift) & 255u], 1u);
        }
        __syncthreads();
        if (tid == 0) {
            unsigned kw = s_kwant, cum = 0;
            for (int b = 255; b >= 0; --b) {
                unsigned c = hist[b];
                if (cum + c >= kw) {
                    s_prefix = prefix | ((unsigned)b << shift);
                    s_kwant  = kw - cum;
                    break;
                }
                cum += c;
            }
        }
        __syncthreads();
    }

    const unsigned T = s_prefix;            // key of k-th largest
    const unsigned ntake = s_kwant;         // #ties (==T) to include

    unsigned tc = 0;
    for (int i = 0; i < 32; ++i) tc += (keys[base + i] == T) ? 1u : 0u;
    scn[tid + 1] = tc;
    if (tid == 0) scn[0] = 0;
    __syncthreads();
    if (tid == 0) for (int t = 1; t <= 256; ++t) scn[t] += scn[t - 1];
    __syncthreads();

    unsigned trank = scn[tid];
    for (int i = 0; i < 32; ++i) {
        unsigned key = keys[base + i];
        bool sel;
        if (key > T)       sel = true;
        else if (key == T) { sel = (trank < ntake); ++trank; }
        else               sel = false;
        unsigned u = (key & 0x80000000u) ? (key ^ 0x80000000u) : ~key;
        float f = __uint_as_float(u);
        p[base + i] = sel ? fmaxf(f, 0.0f) : 0.0f;
    }
}

// ---------------------------------------------------------------------------
// W_dec [D,H] -> WdT [H,D] (L2-resident, enables coalesced sparse decode)
// ---------------------------------------------------------------------------
__global__ __launch_bounds__(256)
void transpose_kernel(const float* __restrict__ W, float* __restrict__ WT) {
    __shared__ float tile[32][33];
    const int tx = threadIdx.x & 31;
    const int ty = threadIdx.x >> 5;        // 0..7
    const int h0 = blockIdx.x * 32;
    const int d0 = blockIdx.y * 32;
    #pragma unroll
    for (int j = 0; j < 4; ++j)
        tile[ty + j * 8][tx] = W[(size_t)(d0 + ty + j * 8) * D_HIDDEN + h0 + tx];
    __syncthreads();
    #pragma unroll
    for (int j = 0; j < 4; ++j)
        WT[(size_t)(h0 + ty + j * 8) * D_MODEL + d0 + tx] = tile[tx][ty + j * 8];
}

// ---------------------------------------------------------------------------
// Sparse decode + per-row squared-error partial + l0 count.
// One block per batch row; compacts <=64 positive activations into LDS with
// deterministic index-ordered positions, then gathers W rows.
// ---------------------------------------------------------------------------
__global__ __launch_bounds__(256)
void decode_kernel(const float* __restrict__ hidden,
                   const float* __restrict__ x,
                   const float* __restrict__ Wsrc,   // WdT[H,D] if transposed else W[D,H]
                   const float* __restrict__ b_dec,
                   const float* __restrict__ b_pre,
                   float* __restrict__ rec,
                   float* __restrict__ partial,
                   unsigned* __restrict__ l0_count,
                   int transposed) {
    const int row = blockIdx.x;
    const int tid = threadIdx.x;
    const float* hrow = hidden + (size_t)row * D_HIDDEN;

    __shared__ int   sidx[TOPK];
    __shared__ float sval[TOPK];
    __shared__ int   scn[257];
    __shared__ float red[256];

    const int cbase = tid * 32;
    int cnt = 0;
    for (int i = 0; i < 32; ++i) cnt += (hrow[cbase + i] > 0.0f) ? 1 : 0;
    scn[tid + 1] = cnt;
    if (tid == 0) scn[0] = 0;
    __syncthreads();
    if (tid == 0) for (int t = 1; t <= 256; ++t) scn[t] += scn[t - 1];
    __syncthreads();
    int pos = scn[tid];
    for (int i = 0; i < 32; ++i) {
        float v = hrow[cbase + i];
        if (v > 0.0f) { sidx[pos] = cbase + i; sval[pos] = v; ++pos; }
    }
    __syncthreads();
    const int total = scn[256];

    float acc[8];
    #pragma unroll
    for (int q = 0; q < 8; ++q) {
        int d = tid + q * 256;
        acc[q] = b_dec[d] + b_pre[d];
    }
    for (int i = 0; i < total; ++i) {
        const int   h = sidx[i];
        const float v = sval[i];
        if (transposed) {
            const float* wr = Wsrc + (size_t)h * D_MODEL;
            #pragma unroll
            for (int q = 0; q < 8; ++q) acc[q] = fmaf(v, wr[tid + q * 256], acc[q]);
        } else {
            #pragma unroll
            for (int q = 0; q < 8; ++q)
                acc[q] = fmaf(v, Wsrc[(size_t)(tid + q * 256) * D_HIDDEN + h], acc[q]);
        }
    }

    float ss = 0.0f;
    const float* xrow = x   + (size_t)row * D_MODEL;
    float*       rrow = rec + (size_t)row * D_MODEL;
    #pragma unroll
    for (int q = 0; q < 8; ++q) {
        int d = tid + q * 256;
        rrow[d] = acc[q];
        float df = acc[q] - xrow[d];
        ss = fmaf(df, df, ss);
    }
    red[tid] = ss;
    __syncthreads();
    for (int s = 128; s > 0; s >>= 1) {
        if (tid < s) red[tid] += red[tid + s];
        __syncthreads();
    }
    if (tid == 0) {
        partial[row] = red[0];
        atomicAdd(l0_count, (unsigned)total);   // integer: deterministic
    }
}

__global__ void init_kernel(unsigned* c) { if (threadIdx.x == 0) *c = 0u; }

__global__ __launch_bounds__(256)
void finalize_kernel(const float* __restrict__ partial,
                     const unsigned* __restrict__ l0_count,
                     float* __restrict__ out_scalars) {
    __shared__ float red[256];
    const int tid = threadIdx.x;
    float s = 0.0f;
    for (int i = 0; i < 32; ++i) s += partial[tid * 32 + i];  // fixed order
    red[tid] = s;
    __syncthreads();
    for (int st = 128; st > 0; st >>= 1) {
        if (tid < st) red[tid] += red[tid + st];
        __syncthreads();
    }
    if (tid == 0) {
        float rl = red[0] / ((float)BATCH * (float)D_MODEL);
        out_scalars[0] = rl;                                   // loss
        out_scalars[1] = rl;                                   // reconstruction_loss
        out_scalars[2] = 0.0f;                                 // sparsity_loss
        out_scalars[3] = (float)(*l0_count) / (float)BATCH;    // l0
    }
}

extern "C" void kernel_launch(void* const* d_in, const int* in_sizes, int n_in,
                              void* d_out, int out_size, void* d_ws, size_t ws_size,
                              hipStream_t stream) {
    const float* x     = (const float*)d_in[0];
    const float* W_enc = (const float*)d_in[1];
    const float* b_enc = (const float*)d_in[2];
    const float* W_dec = (const float*)d_in[3];
    const float* b_dec = (const float*)d_in[4];
    const float* b_pre = (const float*)d_in[5];
    // d_in[6] = k (compile-time TOPK)

    float* rec     = (float*)d_out;
    float* hidden  = rec + (size_t)BATCH * D_MODEL;
    float* scalars = hidden + (size_t)BATCH * D_HIDDEN;

    unsigned char* ws = (unsigned char*)d_ws;
    unsigned* l0_cnt = (unsigned*)ws;                    // [0,4)
    float*    partial = (float*)(ws + 1024);             // 32 KB
    float*    WdT     = (float*)(ws + 65536);            // 64 MB optional
    const size_t need_T = 65536 + (size_t)D_MODEL * D_HIDDEN * sizeof(float);
    const int use_T = (ws_size >= need_T) ? 1 : 0;

    init_kernel<<<1, 32, 0, stream>>>(l0_cnt);

    dim3 egrid(D_HIDDEN / 128, BATCH / 128);
    encode_kernel<<<egrid, 256, 0, stream>>>(x, W_enc, b_enc, b_pre, hidden);

    topk_kernel<<<BATCH, 256, 0, stream>>>(hidden);

    if (use_T) {
        dim3 tgrid(D_HIDDEN / 32, D_MODEL / 32);
        transpose_kernel<<<tgrid, 256, 0, stream>>>(W_dec, WdT);
    }

    decode_kernel<<<BATCH, 256, 0, stream>>>(hidden, x, use_T ? WdT : W_dec,
                                             b_dec, b_pre, rec, partial, l0_cnt, use_T);

    finalize_kernel<<<1, 256, 0, stream>>>(partial, l0_cnt, scalars);
}
